// MultiHeadAttention_29652454211670
// MI455X (gfx1250) — matrix operand order repaired and checked by reading, not compiled, not run on hardware
//
#include <hip/hip_runtime.h>

// ---------------------------------------------------------------------------
// MI455X (gfx1250, wave32) multi-head attention with KV cache.
// All GEMMs via v_wmma_f32_16x16x32_bf16 (f32 accumulate), LDS-staged tiles.
// f32->bf16 conversions use native packed converts (__builtin_convertvector).
// ---------------------------------------------------------------------------

typedef __attribute__((ext_vector_type(16))) __bf16 v16bf;
typedef __attribute__((ext_vector_type(8)))  __bf16 v8bf;
typedef __attribute__((ext_vector_type(8)))  float  v8fv;
typedef __attribute__((ext_vector_type(8)))  float  v8f;
typedef unsigned short u16_t;

struct BfFrag { uint4 lo, hi; };

__device__ __forceinline__ v16bf make_frag(uint4 lo, uint4 hi) {
  BfFrag f{lo, hi};
  return __builtin_bit_cast(v16bf, f);
}

__device__ __forceinline__ u16_t f2bf(float f) {
  return __builtin_bit_cast(u16_t, (__bf16)f);   // native cvt, RNE
}

#define BM 128
#define BN 128
#define BK 32
#define LDSS 40   // row stride in bf16 elements (80B, multiple of 16B)

// C = A * B^T (+bias) (*scale) (mask) ; A: [M,K] (f32, cvt->bf16 on load),
// B: [Nn,K] row-major bf16 (i.e. weights [out,in] or K-cache [s,d] or V^T [d,s]).
// Batched over blockIdx.z = n*16+h with separate n/h strides.
template<int A_F32, int USE_BIAS, int USE_MASK, int STORE_KV>
__global__ __launch_bounds__(256) void gemm_wmma(
    const void* __restrict__ A_, const u16_t* __restrict__ B_,
    const float* __restrict__ bias, float* __restrict__ C_,
    const int* __restrict__ mask_,
    int M, int Nn, int K, int lda, int ldb, int ldc,
    long sAn, long sAh, long sBn, long sBh, long sCn, long sCh, long sMn,
    float scale)
{
  __shared__ u16_t As[BM * LDSS];
  __shared__ u16_t Bs[BN * LDSS];

  const int tid   = threadIdx.x;
  const int lane  = tid & 31;
  const int wave  = tid >> 5;     // 0..7
  const int waveM = wave >> 2;    // 0..1  -> 64 rows each
  const int waveN = wave & 3;     // 0..3  -> 32 cols each
  const int half  = lane >> 4;    // 0/1
  const int l16   = lane & 15;

  const int z  = blockIdx.z;
  const int zn = z >> 4, zh = z & 15;

  const int blockM = blockIdx.y * BM;
  const int blockN = blockIdx.x * BN;

  const u16_t* Bp = B_ + (size_t)zn * sBn + (size_t)zh * sBh;
  float*       Cp = C_ + (size_t)zn * sCn + (size_t)zh * sCh;
  const int*   maskp = USE_MASK ? (mask_ + (size_t)zn * sMn) : nullptr;

  // tile loader indices: 256 threads, each moves 16 bf16 (32B) per tile
  const int lrow = tid >> 1;          // 0..127
  const int lcol = (tid & 1) * 16;    // 0 or 16

  v8f acc[4][2] = {};

  for (int k0 = 0; k0 < K; k0 += BK) {
    // ---- stage A tile [BM x BK] into LDS as bf16 ----
    if (A_F32) {
      const float* Ap  = (const float*)A_ + (size_t)zn * sAn + (size_t)zh * sAh;
      const float* src = Ap + (size_t)(blockM + lrow) * lda + k0 + lcol;
      v8fv f0 = *(const v8fv*)(src);       // 32B (addresses are 32B aligned)
      v8fv f1 = *(const v8fv*)(src + 8);
      v8bf b0 = __builtin_convertvector(f0, v8bf);   // packed cvt, RNE
      v8bf b1 = __builtin_convertvector(f1, v8bf);
      u16_t* d = &As[lrow * LDSS + lcol];
      *(uint4*)(d)     = __builtin_bit_cast(uint4, b0);
      *(uint4*)(d + 8) = __builtin_bit_cast(uint4, b1);
      if (k0 + BK < K) __builtin_prefetch(src + BK, 0, 1);   // global_prefetch_b8
    } else {
      const u16_t* Ap  = (const u16_t*)A_ + (size_t)zn * sAn + (size_t)zh * sAh;
      const u16_t* src = Ap + (size_t)(blockM + lrow) * lda + k0 + lcol;
      uint4 u0 = *(const uint4*)(src);
      uint4 u1 = *(const uint4*)(src + 8);
      u16_t* d = &As[lrow * LDSS + lcol];
      *(uint4*)(d)     = u0;
      *(uint4*)(d + 8) = u1;
      if (k0 + BK < K) __builtin_prefetch(src + BK, 0, 1);
    }
    // ---- stage B tile [BN x BK] (rows = output columns) ----
    {
      const u16_t* src = Bp + (size_t)(blockN + lrow) * ldb + k0 + lcol;
      uint4 u0 = *(const uint4*)(src);
      uint4 u1 = *(const uint4*)(src + 8);
      u16_t* d = &Bs[lrow * LDSS + lcol];
      *(uint4*)(d)     = u0;
      *(uint4*)(d + 8) = u1;
      if (k0 + BK < K) __builtin_prefetch(src + BK, 0, 1);
    }
    __syncthreads();

    // ---- fragments per ISA 16-bit A/B layouts ----
    const int abase = half * 8;     // A: K chunks {abase..+7, abase+16..+23}
    v16bf afrag[4], bfrag[2];
    #pragma unroll
    for (int i = 0; i < 4; ++i) {
      const u16_t* p = &As[(waveM * 64 + i * 16 + l16) * LDSS];
      afrag[i] = make_frag(*(const uint4*)(p + abase),
                           *(const uint4*)(p + abase + 16));
    }
    #pragma unroll
    for (int j = 0; j < 2; ++j) {
      const u16_t* p = &Bs[(waveN * 32 + j * 16 + l16) * LDSS + half * 8];
      bfrag[j] = make_frag(*(const uint4*)(p), *(const uint4*)(p + 16));
    }
    #pragma unroll
    for (int i = 0; i < 4; ++i)
      #pragma unroll
      for (int j = 0; j < 2; ++j)
        acc[i][j] = __builtin_amdgcn_wmma_f32_16x16x32_bf16(
            false, afrag[i], false, bfrag[j], (short)0, acc[i][j], false, false);

    __syncthreads();
  }

  // ---- epilogue: C layout VGPR r -> M = r + 8*half, N = l16 ----
  #pragma unroll
  for (int i = 0; i < 4; ++i) {
    const int row0 = blockM + waveM * 64 + i * 16 + half * 8;
    #pragma unroll
    for (int j = 0; j < 2; ++j) {
      const int col = blockN + waveN * 32 + j * 16 + l16;
      const float b = USE_BIAS ? bias[col] : 0.0f;
      #pragma unroll
      for (int r = 0; r < 8; ++r) {
        const int row = row0 + r;
        float v = acc[i][j][r] * scale + b;
        if (USE_MASK) {
          if (maskp[(long)row * Nn + col] == 0) v = -1000000000.0f;
        }
        if (STORE_KV) {
          // row = n*1024 + t ; col = h*128 + d -> past[n,h,t,d] (S-major 2048)
          const int n = row >> 10, t = row & 1023, h = col >> 7, dd = col & 127;
          Cp[(((long)(n * 16 + h) * 2048 + t) << 7) + dd] = v;
        } else {
          Cp[(long)row * ldc + col] = v;
        }
      }
    }
  }
}

// ---------------- elementwise helpers ----------------

__global__ void k_f32_to_bf16(const float* __restrict__ s,
                              u16_t* __restrict__ d, int n) {
  int i = blockIdx.x * blockDim.x + threadIdx.x;
  if (i < n) d[i] = f2bf(s[i]);
}

// v[n,h,s,d] f32 -> vT[n,h,d,s] bf16   (8,388,608 elements)
__global__ void k_transpose_v(const float* __restrict__ v,
                              u16_t* __restrict__ vt) {
  int i = blockIdx.x * blockDim.x + threadIdx.x;
  int nh  = i >> 18;          // / (128*2048)
  int rem = i & 262143;
  int d   = rem >> 11;        // / 2048
  int s   = rem & 2047;
  vt[i] = f2bf(v[((size_t)nh << 18) + ((size_t)s << 7) + d]);
}

// past_{k,v}[n,h,p,d] -> out past region rows s in [1024,2048)
__global__ void k_copy_past(const float* __restrict__ p,
                            float* __restrict__ dst) {
  int i = blockIdx.x * blockDim.x + threadIdx.x;   // 4,194,304
  int nh = i >> 17;           // / (1024*128)
  int w  = i & 131071;
  dst[((size_t)nh << 18) + 131072 + w] = p[i];
}

// softmax over the HEAD axis: for each (n,t,s) normalize 16 values
// strided by T*S = 2,097,152 within scores[n,h,t,s].
__global__ void k_softmax_head(float* __restrict__ sc) {
  int i = blockIdx.x * blockDim.x + threadIdx.x;   // 4,194,304
  int n   = i >> 21;
  int rem = i & 2097151;
  float* p = sc + ((size_t)n << 25) + rem;
  float v[16];
  float m = -3.0e38f;
  #pragma unroll
  for (int h = 0; h < 16; ++h) { v[h] = p[(size_t)h << 21]; m = fmaxf(m, v[h]); }
  float sum = 0.0f;
  #pragma unroll
  for (int h = 0; h < 16; ++h) { v[h] = __expf(v[h] - m); sum += v[h]; }
  const float inv = 1.0f / sum;
  #pragma unroll
  for (int h = 0; h < 16; ++h) p[(size_t)h << 21] = v[h] * inv;
}

// ---------------------------------------------------------------------------

extern "C" void kernel_launch(void* const* d_in, const int* in_sizes, int n_in,
                              void* d_out, int out_size, void* d_ws, size_t ws_size,
                              hipStream_t stream) {
  (void)in_sizes; (void)n_in; (void)out_size; (void)ws_size;
  const float* q      = (const float*)d_in[0];
  const float* k      = (const float*)d_in[1];
  const float* v      = (const float*)d_in[2];
  const float* past_k = (const float*)d_in[3];
  const float* past_v = (const float*)d_in[4];
  const int*   mask   = (const int*)  d_in[5];
  const float* Wq = (const float*)d_in[6];   const float* bq = (const float*)d_in[7];
  const float* Wk = (const float*)d_in[8];   const float* bk = (const float*)d_in[9];
  const float* Wv = (const float*)d_in[10];  const float* bv = (const float*)d_in[11];
  const float* Wo = (const float*)d_in[12];  const float* bo = (const float*)d_in[13];
  float* out = (float*)d_out;

  const long HH = 2048L * 2048L;             // 4,194,304
  const long KV = 2L * 16 * 2048 * 128;      // 8,388,608 (per k or v cache)

  // workspace carve (needs ~352 MB)
  char* ws = (char*)d_ws;
  size_t off = 0;
  auto carve = [&](size_t bytes) -> void* {
    void* p = ws + off; off += (bytes + 255) & ~(size_t)255; return p;
  };
  u16_t* Wqb    = (u16_t*)carve(HH * 2);
  u16_t* Wkb    = (u16_t*)carve(HH * 2);
  u16_t* Wvb    = (u16_t*)carve(HH * 2);
  u16_t* Wob    = (u16_t*)carve(HH * 2);
  float* qh     = (float*)carve(HH * 4);                 // [n,t,(h,d)]
  u16_t* kcb    = (u16_t*)carve(KV * 2);                 // kcat bf16 [n,h,s,d]
  u16_t* vtb    = (u16_t*)carve(KV * 2);                 // vT  bf16 [n,h,d,s]
  float* scores = (float*)carve(2L*16*1024*2048 * 4);    // [n,h,t,s]
  float* aout   = (float*)carve(2L*16*1024*128 * 4);     // [n,h,t,d]

  float* pastK = out + 4194304;              // past[0]: [n,h,s,d]
  float* pastV = out + 4194304 + 8388608;    // past[1]

  const dim3 blk(256);

  // weights -> bf16
  k_f32_to_bf16<<<dim3((unsigned)(HH / 256)), blk, 0, stream>>>(Wq, Wqb, (int)HH);
  k_f32_to_bf16<<<dim3((unsigned)(HH / 256)), blk, 0, stream>>>(Wk, Wkb, (int)HH);
  k_f32_to_bf16<<<dim3((unsigned)(HH / 256)), blk, 0, stream>>>(Wv, Wvb, (int)HH);
  k_f32_to_bf16<<<dim3((unsigned)(HH / 256)), blk, 0, stream>>>(Wo, Wob, (int)HH);

  // QKV projections: [2048,2048] = act[2048,2048] * W^T + b
  const dim3 gproj(16, 16, 1);
  gemm_wmma<1,1,0,0><<<gproj, blk, 0, stream>>>(q, Wqb, bq, qh, nullptr,
      2048, 2048, 2048, 2048, 2048, 2048, 0,0, 0,0, 0,0, 0, 1.0f);
  gemm_wmma<1,1,0,1><<<gproj, blk, 0, stream>>>(k, Wkb, bk, pastK, nullptr,
      2048, 2048, 2048, 2048, 2048, 0,    0,0, 0,0, 0,0, 0, 1.0f);
  gemm_wmma<1,1,0,1><<<gproj, blk, 0, stream>>>(v, Wvb, bv, pastV, nullptr,
      2048, 2048, 2048, 2048, 2048, 0,    0,0, 0,0, 0,0, 0, 1.0f);

  // append cache: s in [1024,2048)
  k_copy_past<<<dim3(16384), blk, 0, stream>>>(past_k, pastK);
  k_copy_past<<<dim3(16384), blk, 0, stream>>>(past_v, pastV);

  // kcat -> bf16 (same layout), vcat -> bf16 transposed [n,h,d,s]
  k_f32_to_bf16<<<dim3(32768), blk, 0, stream>>>(pastK, kcb, (int)KV);
  k_transpose_v<<<dim3(32768), blk, 0, stream>>>(pastV, vtb);

  // scores[n,h,t,s] = (qh . kcat^T)/sqrt(D), mask==0 -> -1e9 ; batched z = n*16+h
  const dim3 gsc(16, 8, 32);
  gemm_wmma<1,0,1,0><<<gsc, blk, 0, stream>>>(qh, kcb, nullptr, scores, mask,
      1024, 2048, 128, 2048, 128, 2048,
      2097152L, 128L,        // A: n-> T*HID, h-> 128 cols
      4194304L, 262144L,     // B: n-> 16*S*D, h-> S*D
      33554432L, 2097152L,   // C: n-> 16*T*S, h-> T*S
      2097152L,              // mask: n-> T*S
      0.08838834764831845f);

  // softmax over head axis, in place
  k_softmax_head<<<dim3(16384), blk, 0, stream>>>(scores);

  // aout[n,h,t,d] = attn . vcat ; B = vT[n,h,d,s]
  const dim3 gav(1, 8, 32);
  gemm_wmma<1,0,0,0><<<gav, blk, 0, stream>>>(scores, vtb, nullptr, aout, nullptr,
      1024, 128, 2048, 2048, 2048, 128,
      33554432L, 2097152L,   // A strides
      4194304L, 262144L,     // B strides
      2097152L, 131072L,     // C strides
      0, 1.0f);

  // x = reshape(aout) @ Wo^T + bo   (raw reshape == contiguous reinterpret)
  gemm_wmma<1,1,0,0><<<gproj, blk, 0, stream>>>(aout, Wob, bo, out, nullptr,
      2048, 2048, 2048, 2048, 2048, 2048, 0,0, 0,0, 0,0, 0, 1.0f);
}
